// SingleRenderer_23433341567115
// MI455X (gfx1250) — matrix operand order repaired and checked by reading, not compiled, not run on hardware
//
#include <hip/hip_runtime.h>

#define R_RAYS   2048
#define N_CRS    128
#define N_IMP    64
#define N_TOT    192
#define WW       256
#define LDH      264   // padded row stride (halves) for 256-wide activations
#define LDHIN    288   // padded radiance input width (265 -> 288 = 9*32)

typedef __attribute__((ext_vector_type(16))) _Float16 v16h;
typedef __attribute__((ext_vector_type(8)))  _Float16 v8h;
typedef __attribute__((ext_vector_type(8)))  float    v8f;

__device__ __forceinline__ float softplusf(float x) {
    return (x > 20.f) ? x : log1pf(expf(x));
}
__device__ __forceinline__ float laplace_cdf(float x, float beta) {
    float ib = 1.f / beta;
    return (x <= 0.f) ? 0.5f * expf(x * ib) : 1.f - 0.5f * expf(-x * ib);
}
__device__ __forceinline__ float sigmoid_f(float x) { return 1.f / (1.f + expf(-x)); }

// ---- WMMA fragment loaders (contiguous k order, same in A and B) ----------------
// A (16x32 f16): lanes0-15 row=lane, K in {k0..k0+15}, k0 = kb + 16*(lane>=16)
__device__ __forceinline__ v16h load_frag_A(const _Float16* lds, int ldh, int mBase, int kBase) {
    int lane = threadIdx.x & 31;
    int row  = mBase + (lane & 15);
    int k0   = kBase + ((lane >> 4) << 4);
    const _Float16* p = lds + row * ldh + k0;
    v8h lo = *(const v8h*)p;
    v8h hi = *(const v8h*)(p + 8);
    v16h a;
#pragma unroll
    for (int i = 0; i < 8; ++i) { a[i] = lo[i]; a[8 + i] = hi[i]; }
    return a;
}
// B (32x16 f16) from COLUMN-MAJOR weights Wc[n*ldK + k]:
// lane: n = nBase + (lane&15); halves h: K = kBase + 16*(lane>=16) + h  -> 32 contiguous bytes
__device__ __forceinline__ v16h load_frag_B(const _Float16* Wc, int ldK, int kBase, int nBase) {
    int lane = threadIdx.x & 31;
    int n    = nBase + (lane & 15);
    int k0   = kBase + ((lane >> 4) << 4);
    const _Float16* p = Wc + (size_t)n * ldK + k0;
    v8h lo = *(const v8h*)p;
    v8h hi = *(const v8h*)(p + 8);
    v16h b;
#pragma unroll
    for (int i = 0; i < 8; ++i) { b[i] = lo[i]; b[8 + i] = hi[i]; }
    return b;
}

// ---- 64x256 = (64xK) * (Kx256) GEMM, 256 threads (8 waves), WMMA core ----
// wave w owns N columns [32w, 32w+32); 4 M-tiles of 16 -> 8 WMMAs per k-step.
// Epilogue epi(m, n, val) applied per output element; safe for in-place use.
template <int KSTEPS, class Epi>
__device__ __forceinline__ void gemm64(const _Float16* actLds, int ldA,
                                       const _Float16* Wc, int ldK, Epi epi) {
    int wave  = threadIdx.x >> 5;
    int nBase = wave * 32;
    v8f acc[4][2];
#pragma unroll
    for (int i = 0; i < 4; ++i)
#pragma unroll
        for (int j = 0; j < 2; ++j)
#pragma unroll
            for (int r = 0; r < 8; ++r) acc[i][j][r] = 0.f;

#pragma unroll
    for (int ks = 0; ks < KSTEPS; ++ks) {
        int kb = ks * 32;
        v16h b0 = load_frag_B(Wc, ldK, kb, nBase);
        v16h b1 = load_frag_B(Wc, ldK, kb, nBase + 16);
#pragma unroll
        for (int mt = 0; mt < 4; ++mt) {
            v16h a = load_frag_A(actLds, ldA, mt * 16, kb);
            acc[mt][0] = __builtin_amdgcn_wmma_f32_16x16x32_f16(false, a, false, b0,
                                                               (short)0, acc[mt][0], false, false);
            acc[mt][1] = __builtin_amdgcn_wmma_f32_16x16x32_f16(false, a, false, b1,
                                                               (short)0, acc[mt][1], false, false);
        }
    }
    __syncthreads();  // all reads of actLds complete before epilogue writes (in-place safe)
    int lane = threadIdx.x & 31;
    int nIn  = lane & 15;
    int mOff = (lane >> 4) << 3;
#pragma unroll
    for (int mt = 0; mt < 4; ++mt)
#pragma unroll
        for (int nt = 0; nt < 2; ++nt)
#pragma unroll
            for (int r = 0; r < 8; ++r)
                epi(mt * 16 + mOff + r, nBase + nt * 16 + nIn, acc[mt][nt][r]);
    __syncthreads();
}

// ---- prep: convert weights to f16, column-major (and transposes for bwd) --
__global__ void prep_kernel(const float* w1, const float* w2, const float* w3,
                            const float* rw0, const float* rw1,
                            _Float16* w1c, _Float16* w2c, _Float16* w1bc, _Float16* w2bc,
                            _Float16* w3fc, _Float16* rw0c, _Float16* rw1c) {
    int i0 = blockIdx.x * blockDim.x + threadIdx.x;
    int stride = gridDim.x * blockDim.x;
    for (int i = i0; i < WW * WW; i += stride) {
        int n = i >> 8, k = i & 255;
        w1c[i]  = (_Float16)w1[k * WW + n];        // forward B: W^T col-major
        w2c[i]  = (_Float16)w2[k * WW + n];
        w1bc[i] = (_Float16)w1[i];                 // backward B: (W^T) col-major == W row-major
        w2bc[i] = (_Float16)w2[i];
        w3fc[i] = (_Float16)w3[k * 257 + 1 + n];   // feature cols of sdf_w3
        rw1c[i] = (_Float16)rw1[k * WW + n];
    }
    for (int i = i0; i < WW * LDHIN; i += stride) {   // rw0 padded 265 -> 288 rows
        int n = i / LDHIN, k = i - n * LDHIN;
        rw0c[i] = (k < 265) ? (_Float16)rw0[k * WW + n] : (_Float16)0.f;
    }
}

// ---- coarse pass: sdf -> sigma for R*128 points ---------------------------
__global__ void __launch_bounds__(256) coarse_kernel(
    const float* rays_o, const float* rays_d, const float* alpha, const float* beta,
    const float* w0, const float* b0, const float* b1, const float* b2,
    const float* w3, const float* b3,
    const _Float16* w1c, const _Float16* w2c, float* sig_out) {
    __shared__ __align__(16) _Float16 hbuf[64 * LDH];
    __shared__ float px[64], py[64], pz[64];
    __shared__ float w3c0[WW];
    int tid = threadIdx.x;
    int base = blockIdx.x * 64;
    w3c0[tid] = w3[tid * 257];
    if (tid < 64) {
        int idx = base + tid;
        int ray = idx / N_CRS;
        int s   = idx - ray * N_CRS;
        float ox = rays_o[ray * 3], oy = rays_o[ray * 3 + 1], oz = rays_o[ray * 3 + 2];
        float dx = rays_d[ray * 3], dy = rays_d[ray * 3 + 1], dz = rays_d[ray * 3 + 2];
        float inv = rsqrtf(dx * dx + dy * dy + dz * dz);
        dx *= inv; dy *= inv; dz *= inv;
        float d = 6.0f * (float)s / 127.0f;
        px[tid] = ox + dx * d; py[tid] = oy + dy * d; pz[tid] = oz + dz * d;
    }
    __syncthreads();
    for (int e = tid; e < 64 * WW; e += 256) {   // layer0 (K=3) on VALU
        int p = e >> 8, j = e & 255;
        float z = px[p] * w0[j] + py[p] * w0[256 + j] + pz[p] * w0[512 + j] + b0[j];
        hbuf[p * LDH + j] = (_Float16)softplusf(z);
    }
    __syncthreads();
    gemm64<8>(hbuf, LDH, w1c, 256, [&](int m, int n, float v) {
        hbuf[m * LDH + n] = (_Float16)softplusf(v + b1[n]);
    });
    gemm64<8>(hbuf, LDH, w2c, 256, [&](int m, int n, float v) {
        hbuf[m * LDH + n] = (_Float16)softplusf(v + b2[n]);
    });
    if (tid < 64) {
        float s = b3[0];
        for (int k = 0; k < WW; ++k) s += (float)hbuf[tid * LDH + k] * w3c0[k];
        sig_out[base + tid] = alpha[0] * laplace_cdf(-s, beta[0]);
    }
}

// ---- hierarchical importance sampling + merge sort (1 thread / ray) -------
__global__ void sampler_kernel(const float* sig, float* dall) {
    int r = blockIdx.x * blockDim.x + threadIdx.x;
    if (r >= R_RAYS) return;
    const float delta = 6.0f / 127.0f;
    float w[127], cdf[128], dfine[64];
    float T = 1.f, wsum = 0.f;
    for (int i = 0; i < 127; ++i) {
        float a  = 1.f - expf(-sig[r * N_CRS + i] * delta);
        float wi = a * T;
        T *= (1.f - a + 1e-10f);
        w[i] = wi + 1e-5f;
        wsum += w[i];
    }
    float inv = 1.f / wsum;
    cdf[0] = 0.f;
    for (int i = 0; i < 127; ++i) cdf[i + 1] = cdf[i] + w[i] * inv;
    for (int j = 0; j < 64; ++j) {
        float u = (float)j / 63.0f;
        int ind = 128;
        for (int i = 0; i < 128; ++i) { if (cdf[i] >= u) { ind = i; break; } }
        int below = ind - 1; if (below < 0) below = 0; if (below > 127) below = 127;
        int above = ind;     if (above > 127) above = 127;
        float cb = cdf[below], ca = cdf[above];
        float bb = 6.0f * (float)below / 127.0f;
        float ba = 6.0f * (float)above / 127.0f;
        float den = (ca - cb < 1e-5f) ? 1.f : (ca - cb);
        dfine[j] = bb + (u - cb) / den * (ba - bb);
    }
    int ci = 0, fi = 0;
    for (int o = 0; o < N_TOT; ++o) {   // merge two sorted lists
        float dc = 6.0f * (float)ci / 127.0f;
        bool takeC = (fi >= 64) || (ci < 128 && dc <= dfine[fi]);
        if (takeC) { dall[r * N_TOT + o] = dc; ++ci; }
        else       { dall[r * N_TOT + o] = dfine[fi]; ++fi; }
    }
}

// ---- fine pass: sdf fwd + bwd (normals) + radiance, fused per 64-pt tile --
__global__ void __launch_bounds__(256) fine_kernel(
    const float* rays_o, const float* rays_d, const float* alpha, const float* beta,
    const float* w0, const float* b0, const float* b1, const float* b2,
    const float* w3, const float* b3,
    const float* rb0, const float* rb1, const float* rw2, const float* rb2,
    const _Float16* w1c, const _Float16* w2c,
    const _Float16* w1bc, const _Float16* w2bc, const _Float16* w3fc,
    const _Float16* rw0c, const _Float16* rw1c,
    const float* dall, float* sig_out, float* rgb_out) {
    __shared__ __align__(16) _Float16 h0s[64 * LDH];
    __shared__ __align__(16) _Float16 h1s[64 * LDH];
    __shared__ __align__(16) _Float16 h2s[64 * LDH];
    __shared__ __align__(16) _Float16 hins[64 * LDHIN];
    __shared__ float px[64], py[64], pz[64], vx[64], vy[64], vz[64];
    __shared__ float nab[64 * 3];
    __shared__ float w3c0[WW];
    int tid  = threadIdx.x;
    int base = blockIdx.x * 64;
    for (int e = tid; e < 64 * LDHIN; e += 256) hins[e] = (_Float16)0.f;
    w3c0[tid] = w3[tid * 257];
    if (tid < 64) {
        int idx = base + tid;
        int ray = idx / N_TOT;
        float ox = rays_o[ray * 3], oy = rays_o[ray * 3 + 1], oz = rays_o[ray * 3 + 2];
        float dx = rays_d[ray * 3], dy = rays_d[ray * 3 + 1], dz = rays_d[ray * 3 + 2];
        float inv = rsqrtf(dx * dx + dy * dy + dz * dz);
        dx *= inv; dy *= inv; dz *= inv;
        float d = dall[idx];
        px[tid] = ox + dx * d; py[tid] = oy + dy * d; pz[tid] = oz + dz * d;
        vx[tid] = dx; vy[tid] = dy; vz[tid] = dz;
    }
    __syncthreads();
    for (int e = tid; e < 64 * WW; e += 256) {   // sdf layer0
        int p = e >> 8, j = e & 255;
        float z = px[p] * w0[j] + py[p] * w0[256 + j] + pz[p] * w0[512 + j] + b0[j];
        h0s[p * LDH + j] = (_Float16)softplusf(z);
    }
    __syncthreads();
    // sdf forward
    gemm64<8>(h0s, LDH, w1c, 256, [&](int m, int n, float v) {
        h1s[m * LDH + n] = (_Float16)softplusf(v + b1[n]);
    });
    gemm64<8>(h1s, LDH, w2c, 256, [&](int m, int n, float v) {
        h2s[m * LDH + n] = (_Float16)softplusf(v + b2[n]);
    });
    gemm64<8>(h2s, LDH, w3fc, 256, [&](int m, int n, float v) {   // feat -> radiance input
        hins[m * LDHIN + 9 + n] = (_Float16)(v + b3[1 + n]);
    });
    if (tid < 64) {                                               // sdf scalar + sigma
        float s = b3[0];
        for (int k = 0; k < WW; ++k) s += (float)h2s[tid * LDH + k] * w3c0[k];
        sig_out[base + tid] = alpha[0] * laplace_cdf(-s, beta[0]);
    }
    __syncthreads();
    // backward for normals: softplus'(z) = sigmoid(z) = 1 - exp(-softplus(z)) = 1 - exp(-h)
    for (int e = tid; e < 64 * WW; e += 256) {                    // g2 = w3[:,0] * sp'(z2)
        int p = e >> 8, j = e & 255;
        float h = (float)h2s[p * LDH + j];
        h2s[p * LDH + j] = (_Float16)(w3c0[j] * (1.f - expf(-h)));
    }
    __syncthreads();
    gemm64<8>(h2s, LDH, w2bc, 256, [&](int m, int n, float v) {   // g1 = (g2 @ w2^T) * sp'(z1)
        float h = (float)h1s[m * LDH + n];
        h1s[m * LDH + n] = (_Float16)(v * (1.f - expf(-h)));
    });
    gemm64<8>(h1s, LDH, w1bc, 256, [&](int m, int n, float v) {   // g0 = (g1 @ w1^T) * sp'(z0)
        float h = (float)h0s[m * LDH + n];
        h0s[m * LDH + n] = (_Float16)(v * (1.f - expf(-h)));
    });
    if (tid < 192) {                                              // nabla = g0 @ w0^T (K=256, N=3)
        int p = tid / 3, ax = tid - 3 * p;
        float s = 0.f;
        for (int k = 0; k < WW; ++k) s += (float)h0s[p * LDH + k] * w0[ax * 256 + k];
        nab[p * 3 + ax] = s;
    }
    __syncthreads();
    if (tid < 64) {                                               // normals + radiance input cols
        float nx = nab[tid * 3], ny = nab[tid * 3 + 1], nz = nab[tid * 3 + 2];
        float inv = 1.f / (sqrtf(nx * nx + ny * ny + nz * nz) + 1e-7f);
        _Float16* hi = hins + tid * LDHIN;
        hi[0] = (_Float16)px[tid]; hi[1] = (_Float16)py[tid]; hi[2] = (_Float16)pz[tid];
        hi[3] = (_Float16)vx[tid]; hi[4] = (_Float16)vy[tid]; hi[5] = (_Float16)vz[tid];
        hi[6] = (_Float16)(nx * inv); hi[7] = (_Float16)(ny * inv); hi[8] = (_Float16)(nz * inv);
    }
    __syncthreads();
    // radiance net (K=288 padded, then 256)
    gemm64<9>(hins, LDHIN, rw0c, 288, [&](int m, int n, float v) {
        h0s[m * LDH + n] = (_Float16)fmaxf(v + rb0[n], 0.f);
    });
    gemm64<8>(h0s, LDH, rw1c, 256, [&](int m, int n, float v) {
        h1s[m * LDH + n] = (_Float16)fmaxf(v + rb1[n], 0.f);
    });
    if (tid < 192) {                                              // rgb head (256 -> 3)
        int p = tid / 3, c = tid - 3 * p;
        float s = rb2[c];
        for (int k = 0; k < WW; ++k) s += (float)h1s[p * LDH + k] * rw2[k * 3 + c];
        rgb_out[(base + p) * 3 + c] = sigmoid_f(s);
    }
}

// ---- volume rendering integral (1 thread / ray) ---------------------------
__global__ void accum_kernel(const float* dall, const float* sig, const float* rgb, float* out) {
    int r = blockIdx.x * blockDim.x + threadIdx.x;
    if (r >= R_RAYS) return;
    float T = 1.f, c0 = 0.f, c1 = 0.f, c2 = 0.f;
    for (int i = 0; i < N_TOT; ++i) {
        float d = dall[r * N_TOT + i];
        float delta = (i < N_TOT - 1) ? (dall[r * N_TOT + i + 1] - d) : 1e10f;
        float a = 1.f - expf(-sig[r * N_TOT + i] * delta);
        float w = a * T;
        T *= (1.f - a + 1e-10f);
        c0 += w * rgb[(r * N_TOT + i) * 3 + 0];
        c1 += w * rgb[(r * N_TOT + i) * 3 + 1];
        c2 += w * rgb[(r * N_TOT + i) * 3 + 2];
    }
    out[r * 3 + 0] = c0; out[r * 3 + 1] = c1; out[r * 3 + 2] = c2;
}

extern "C" void kernel_launch(void* const* d_in, const int* in_sizes, int n_in,
                              void* d_out, int out_size, void* d_ws, size_t ws_size,
                              hipStream_t stream) {
    (void)in_sizes; (void)n_in; (void)out_size; (void)ws_size;
    const float* rays_o = (const float*)d_in[0];
    const float* rays_d = (const float*)d_in[1];
    const float* alpha  = (const float*)d_in[2];
    const float* beta   = (const float*)d_in[3];
    const float* w0 = (const float*)d_in[4];  const float* b0 = (const float*)d_in[5];
    const float* w1 = (const float*)d_in[6];  const float* b1 = (const float*)d_in[7];
    const float* w2 = (const float*)d_in[8];  const float* b2 = (const float*)d_in[9];
    const float* w3 = (const float*)d_in[10]; const float* b3 = (const float*)d_in[11];
    const float* rw0 = (const float*)d_in[12]; const float* rb0 = (const float*)d_in[13];
    const float* rw1 = (const float*)d_in[14]; const float* rb1 = (const float*)d_in[15];
    const float* rw2 = (const float*)d_in[16]; const float* rb2 = (const float*)d_in[17];

    char* ws = (char*)d_ws;
    size_t off = 0;
    _Float16* w1c  = (_Float16*)(ws + off); off += (size_t)WW * WW * 2;
    _Float16* w2c  = (_Float16*)(ws + off); off += (size_t)WW * WW * 2;
    _Float16* w1bc = (_Float16*)(ws + off); off += (size_t)WW * WW * 2;
    _Float16* w2bc = (_Float16*)(ws + off); off += (size_t)WW * WW * 2;
    _Float16* w3fc = (_Float16*)(ws + off); off += (size_t)WW * WW * 2;
    _Float16* rw0c = (_Float16*)(ws + off); off += (size_t)WW * LDHIN * 2;
    _Float16* rw1c = (_Float16*)(ws + off); off += (size_t)WW * WW * 2;
    float* sigc = (float*)(ws + off); off += (size_t)R_RAYS * N_CRS * 4;
    float* dall = (float*)(ws + off); off += (size_t)R_RAYS * N_TOT * 4;
    float* sigf = (float*)(ws + off); off += (size_t)R_RAYS * N_TOT * 4;
    float* rgbf = (float*)(ws + off); off += (size_t)R_RAYS * N_TOT * 3 * 4;

    prep_kernel<<<512, 256, 0, stream>>>(w1, w2, w3, rw0, rw1,
                                         w1c, w2c, w1bc, w2bc, w3fc, rw0c, rw1c);
    coarse_kernel<<<R_RAYS * N_CRS / 64, 256, 0, stream>>>(
        rays_o, rays_d, alpha, beta, w0, b0, b1, b2, w3, b3, w1c, w2c, sigc);
    sampler_kernel<<<(R_RAYS + 255) / 256, 256, 0, stream>>>(sigc, dall);
    fine_kernel<<<R_RAYS * N_TOT / 64, 256, 0, stream>>>(
        rays_o, rays_d, alpha, beta, w0, b0, b1, b2, w3, b3,
        rb0, rb1, rw2, rb2, w1c, w2c, w1bc, w2bc, w3fc, rw0c, rw1c,
        dall, sigf, rgbf);
    accum_kernel<<<(R_RAYS + 255) / 256, 256, 0, stream>>>(dall, sigf, rgbf, (float*)d_out);
}
